// GN_21406117003962
// MI455X (gfx1250) — matrix operand order repaired and checked by reading, not compiled, not run on hardware
//
#include <hip/hip_runtime.h>

// ---------------------------------------------------------------------------
// Fused RealNVP coupling-flow kernel for MI455X (gfx1250), bf16 WMMA GEMMs.
//   z:       [65536, 256] f32
//   params:  8 layers x {s,t} nets, each (Wf[256x128], Ws0[128x256],
//            Ws1[256x256], bf[128])  (pytree alphabetical order per net)
//   out:     concat( f[65536,256], logdet[65536], action[65536] ) f32
// ---------------------------------------------------------------------------

typedef __attribute__((ext_vector_type(16))) __bf16          v16bf;
typedef __attribute__((ext_vector_type(16))) unsigned short  v16u;
typedef __attribute__((ext_vector_type(8)))  float           v8f;

union BFVec { v16u u; v16bf bf; };

__device__ __forceinline__ unsigned short f2bf(float x) {
    unsigned int u = __float_as_uint(x);
    u += 0x7FFFu + ((u >> 16) & 1u);           // round-to-nearest-even
    return (unsigned short)(u >> 16);
}

#define LEAKC 0.2f
#define M2C  (-4.0f)
#define LAC   8.0f

// ---------------------------------------------------------------------------
// Weight convert + transpose prologue: f32 row-major -> bf16 [N][K] (K contig)
// ---------------------------------------------------------------------------
__global__ void __launch_bounds__(256) convert_net_kernel(
    const float* __restrict__ W0,   // 128x256
    const float* __restrict__ W1,   // 256x256
    const float* __restrict__ Wf,   // 256x128
    const float* __restrict__ bfv,  // 128
    unsigned short* __restrict__ wt0,  // [256][128]
    unsigned short* __restrict__ wt1,  // [256][256]
    unsigned short* __restrict__ wt2,  // [128][256]
    float* __restrict__ bias)          // [128]
{
    const int i = blockIdx.x * blockDim.x + threadIdx.x;
    if (i < 128 * 256) { const int k = i >> 8, n = i & 255; wt0[n * 128 + k] = f2bf(W0[i]); }
    if (i < 256 * 256) { const int k = i >> 8, n = i & 255; wt1[n * 256 + k] = f2bf(W1[i]); }
    if (i < 256 * 128) { const int k = i >> 7, n = i & 127; wt2[n * 256 + k] = f2bf(Wf[i]); }
    if (i < 128) bias[i] = bfv[i];
}

// ---------------------------------------------------------------------------
// One GEMM stage on a 32-sample tile.
//   A (LDS, bf16 bits): [32][KDIM]
//   WT (global, bf16 bits): [NOUT][KDIM]   (K contiguous)
// 8 waves: wave&1 -> 16-row M strip, wave>>1 -> quarter of N columns.
// ---------------------------------------------------------------------------
template<int KDIM, int NOUT, bool RELU>
__device__ __forceinline__ void gemm_stage(
    const unsigned short* __restrict__ A,
    const unsigned short* __restrict__ WT,
    unsigned short* __restrict__ outBF,   // LDS [32][NOUT] (if RELU)
    float*          __restrict__ outF,    // LDS [32][NOUT] (if !RELU)
    const float*    __restrict__ bias,    // [NOUT]         (if !RELU)
    int wave, int lane)
{
    constexpr int NT = NOUT / 64;                 // 16-col tiles per wave
    const int mBase = (wave & 1) * 16;
    const int nBase = (wave >> 1) * (NOUT / 4);
    const int rowA  = mBase + (lane & 15);
    const int halfA = (lane >> 4) * 16;           // A: lanes 16-31 hold K+16
    const int colL  = lane & 15;
    const int halfB = (lane >> 4) * 16;           // B: lanes 16-31 hold K+16

    v8f acc[NT];
    #pragma unroll
    for (int t = 0; t < NT; ++t)
        #pragma unroll
        for (int r = 0; r < 8; ++r) acc[t][r] = 0.0f;

    #pragma unroll
    for (int kb = 0; kb < KDIM; kb += 32) {
        // A fragment (16x32 bf16), contiguous k order: element s holds k = 16*(lane/16) + s, as for B
        BFVec a;
        const unsigned short* ap = A + rowA * KDIM + kb + halfA;
        #pragma unroll
        for (int j = 0; j < 8; ++j) a.u[j] = ap[j];
        #pragma unroll
        for (int j = 0; j < 8; ++j) a.u[8 + j] = ap[8 + j];

        #pragma unroll
        for (int t = 0; t < NT; ++t) {
            const int n = nBase + t * 16 + colL;
            const unsigned short* bp = WT + n * KDIM + kb + halfB;
            __builtin_prefetch(bp + 64, 0, 0);    // global_prefetch_b8
            BFVec b;
            b.u = *(const v16u*)bp;               // 32B contiguous load
            acc[t] = __builtin_amdgcn_wmma_f32_16x16x32_bf16(
                false, a.bf, false, b.bf, (short)0, acc[t], false, false);
        }
    }

    // Epilogue. C/D layout: VGPR r -> M = r (+8 for lanes 16-31), N = lane&15.
    #pragma unroll
    for (int t = 0; t < NT; ++t) {
        #pragma unroll
        for (int r = 0; r < 8; ++r) {
            const int m = mBase + r + ((lane >> 4) << 3);
            const int n = nBase + t * 16 + colL;
            float v = acc[t][r];
            if constexpr (RELU) {
                v = v > 0.0f ? v : LEAKC * v;
                outBF[m * NOUT + n] = f2bf(v);
            } else {
                v += bias[n];
                outF[m * NOUT + n] = v;
            }
        }
    }
}

// ---------------------------------------------------------------------------
// Fused flow kernel: 1 block = 32 samples, 8 layers + action, field in LDS.
// ---------------------------------------------------------------------------
__global__ void __launch_bounds__(256) flow_fused_kernel(
    const float* __restrict__ z,
    const unsigned short* __restrict__ wts,   // 16 nets x 131072 bf16
    const float* __restrict__ biases,         // 16 x 128 f32
    float* __restrict__ f_out,
    float* __restrict__ d_out_v,
    float* __restrict__ s_out_v)
{
    extern __shared__ char smem[];
    float*          zf    = (float*)smem;                              // 32x256 f32  (32768 B)
    unsigned short* za    = (unsigned short*)(smem + 32768);           // 32x128 u16  ( 8192 B)
    unsigned short* hA    = (unsigned short*)(smem + 40960);           // 32x256 u16  (16384 B)
    unsigned short* hB    = (unsigned short*)(smem + 57344);           // 32x256 u16  (16384 B)
    float*          souts = (float*)(smem + 73728);                    // 32x128 f32  (16384 B)
    float*          touts = (float*)(smem + 90112);                    // 32x128 f32  (16384 B)
    float*          ld    = (float*)(smem + 106496);                   // 32 f32
    float*          sact  = (float*)(smem + 106624);                   // 32 f32
    // total 106752 bytes

    const int tid  = threadIdx.x;
    const int wave = tid >> 5;                 // wave32
    const int lane = tid & 31;
    const int base = blockIdx.x * 32;

    // Load z tile (32x256 f32) with 128-bit loads.
    {
        const float4* zv = (const float4*)(z + (size_t)base * 256);
        float4* zd = (float4*)zf;
        #pragma unroll
        for (int i = 0; i < 8; ++i) zd[tid + i * 256] = zv[tid + i * 256];
    }
    if (tid < 32) { ld[tid] = 0.0f; sact[tid] = 0.0f; }
    __syncthreads();

    #pragma unroll 1
    for (int layer = 0; layer < 8; ++layer) {
        const int p = layer & 1;

        // Gather z_a -> bf16. a-site: row r, col c = 2k + ((r+p)&1).
        for (int e = tid; e < 32 * 128; e += 256) {
            const int smp = e >> 7, j = e & 127;
            const int r = j >> 3, k = j & 7;
            const int c = (k << 1) + ((r + p) & 1);
            za[e] = f2bf(zf[smp * 256 + r * 16 + c]);
        }
        __syncthreads();

        #pragma unroll 1
        for (int net = 0; net < 2; ++net) {     // 0 = s-net, 1 = t-net
            const int netIdx = layer * 2 + net;
            const unsigned short* w0 = wts + (size_t)netIdx * 131072;
            const unsigned short* w1 = w0 + 32768;
            const unsigned short* w2 = w0 + 98304;
            const float* bias = biases + netIdx * 128;
            float* outs = (net == 0) ? souts : touts;

            gemm_stage<128, 256, true >(za, w0, hB, nullptr, nullptr, wave, lane);
            __syncthreads();
            gemm_stage<256, 256, true >(hB, w1, hA, nullptr, nullptr, wave, lane);
            __syncthreads();
            gemm_stage<256, 128, false>(hA, w2, nullptr, outs, bias, wave, lane);
            __syncthreads();
        }

        // f_b = (z_b - t) * exp(-s); logdet += sum(s). 8 threads / sample.
        {
            const int smp  = tid >> 3;
            const int part = tid & 7;
            float ssum = 0.0f;
            #pragma unroll
            for (int jj = 0; jj < 16; ++jj) {
                const int j = part * 16 + jj;
                const int r = j >> 3, k = j & 7;
                const int c = (k << 1) + ((r + p + 1) & 1);   // b-parity
                const int site = r * 16 + c;
                const float s  = souts[smp * 128 + j];
                const float t  = touts[smp * 128 + j];
                const float zb = zf[smp * 256 + site];
                zf[smp * 256 + site] = (zb - t) * __expf(-s);
                ssum += s;
            }
            atomicAdd(&ld[smp], ssum);                        // ds_add_f32
        }
        __syncthreads();
    }

    // phi^4 action with periodic neighbors. 8 threads / sample, 32 sites each.
    {
        const int smp  = tid >> 3;
        const int part = tid & 7;
        const float* f0 = zf + smp * 256;
        float accum = 0.0f;
        #pragma unroll
        for (int ss = 0; ss < 32; ++ss) {
            const int site = part * 32 + ss;
            const int r = site >> 4, c = site & 15;
            const float f = f0[site];
            const float nb = f0[r * 16 + ((c + 1) & 15)]
                           + f0[r * 16 + ((c + 15) & 15)]
                           + f0[(((r + 1) & 15) << 4) + c]
                           + f0[(((r + 15) & 15) << 4) + c];
            const float D = f * ((4.0f + M2C) * f - nb);
            accum += D + LAC * f * f * f * f;
        }
        atomicAdd(&sact[smp], accum);
    }
    __syncthreads();

    // Write outputs.
    {
        float4* fo = (float4*)(f_out + (size_t)base * 256);
        const float4* zs = (const float4*)zf;
        #pragma unroll
        for (int i = 0; i < 8; ++i) fo[tid + i * 256] = zs[tid + i * 256];
        if (tid < 32) {
            d_out_v[base + tid] = ld[tid];
            s_out_v[base + tid] = sact[tid];
        }
    }
}

// ---------------------------------------------------------------------------
extern "C" void kernel_launch(void* const* d_in, const int* in_sizes, int n_in,
                              void* d_out, int out_size, void* d_ws, size_t ws_size,
                              hipStream_t stream)
{
    const float* z = (const float*)d_in[0];

    // Workspace: 16 nets x 131072 bf16 weights, then 16 x 128 f32 biases.
    unsigned short* wts = (unsigned short*)d_ws;
    float* biases = (float*)((char*)d_ws + (size_t)16 * 131072 * sizeof(unsigned short));

    // Input order after z: per layer {s,t}, per net pytree-alphabetical
    // (Wf, Ws[0], Ws[1], bf).
    for (int net = 0; net < 16; ++net) {
        const int layer = net >> 1, st = net & 1;
        const int b = 1 + layer * 8 + st * 4;
        const float* Wf  = (const float*)d_in[b + 0];
        const float* W0  = (const float*)d_in[b + 1];
        const float* W1  = (const float*)d_in[b + 2];
        const float* bfv = (const float*)d_in[b + 3];
        convert_net_kernel<<<256, 256, 0, stream>>>(
            W0, W1, Wf, bfv,
            wts + (size_t)net * 131072,
            wts + (size_t)net * 131072 + 32768,
            wts + (size_t)net * 131072 + 98304,
            biases + net * 128);
    }

    const int batch = in_sizes[0] / 256;
    float* f_out = (float*)d_out;
    float* d_v   = f_out + (size_t)batch * 256;
    float* s_v   = d_v + batch;

    const size_t ldsBytes = 106752;
    flow_fused_kernel<<<dim3(batch / 32), 256, ldsBytes, stream>>>(
        z, wts, biases, f_out, d_v, s_v);
}
